// GraphAttentionClassifier_79912161509621
// MI455X (gfx1250) — matrix operand order repaired and checked by reading, not compiled, not run on hardware
//
#include <hip/hip_runtime.h>

typedef __attribute__((ext_vector_type(16))) _Float16 v16h;
typedef __attribute__((ext_vector_type(8)))  _Float16 v8h;
typedef __attribute__((ext_vector_type(8)))  float    v8f;

#define BATCH  2048
#define NJET   10
#define NLEP   2
#define NHEAD  4
#define E1D    16
#define E2D    32
#define ADIM   128
#define CED    2
#define PFD    7
#define CLD    9
#define NUTJ   45      // C(10,2)
#define HSZ    201     // 12*E1 + CL
#define K1PAD  224     // 201 padded to mult of 32
#define K2PAD  160     // 137 padded to mult of 32
#define EPSF   1e-10f
#define PIF    3.14159265358979323846f

// ---------------- param index map (setup_inputs dict order, leaves flattened) ----
enum ParamIdx {
  P_EMBEDCAT_W = 7, P_EMBEDCAT_B,
  P_EMBEDJETS_W, P_EMBEDJETS_B,
  P_MHAJET_WQKV, P_MHAJET_BQKV, P_MHAJET_WO, P_MHAJET_BO,
  P_CONV_W, P_CONV_B,
  P_EDGEJET_W, P_EDGEJET_B,
  P_MHAEJ_WQKV, P_MHAEJ_BQKV, P_MHAEJ_WO, P_MHAEJ_BO,
  P_EMBEDLEPS_W, P_EMBEDLEPS_B,
  P_MHALEP_WQKV, P_MHALEP_BQKV, P_MHALEP_WO, P_MHALEP_BO,
  P_EDGELEP_W, P_EDGELEP_B,
  P_MHAEL_WQKV, P_MHAEL_BQKV, P_MHAEL_WO, P_MHAEL_BO,
  P_EMBEDLL_W, P_EMBEDLL_B,
  P_EMBEDJL_W, P_EMBEDJL_B,
  P_MHAJL_WQKV, P_MHAJL_BQKV, P_MHAJL_WO, P_MHAJL_BO,
  P_EMBEDJJL_W, P_EMBEDJJL_B,
  P_MHAJJL_WQKV, P_MHAJJL_BQKV, P_MHAJJL_WO, P_MHAJJL_BO,
  P_EMBEDJLL_W, P_EMBEDJLL_B,
  P_MHAJLL_WQKV, P_MHAJLL_BQKV, P_MHAJLL_WO, P_MHAJLL_BO,
  P_EMBEDJJLL_W, P_EMBEDJJLL_B,
  P_MHAJJLL_WQKV, P_MHAJJLL_BQKV, P_MHAJJLL_WO, P_MHAJJLL_BO,
  P_EMBEDGLOB_W, P_EMBEDGLOB_B,
  P_LN_G, P_LN_B,
  P_LN2X_G, P_LN2X_B,
  P_FC1_W, P_FC1_B, P_BN1_G, P_BN1_B, P_BN1_M, P_BN1_V,
  P_FC2_W, P_FC2_B, P_BN2_G, P_BN2_B, P_BN2_M, P_BN2_V,
  P_FC3_W, P_FC3_B, P_BN3_G, P_BN3_B, P_BN3_M, P_BN3_V,
  P_FC4_W, P_FC4_B
};

// ---------------- catfeat one-hot + catembed linear --------------------------
__global__ void catfeat_kernel(const int* __restrict__ cat,
                               const float* __restrict__ Wc, const float* __restrict__ bc,
                               float* __restrict__ catfeat, float* __restrict__ catembed,
                               float* __restrict__ hcat, int B)
{
  int b = blockIdx.x * blockDim.x + threadIdx.x;
  if (b >= B) return;
  float cf[CLD];
  for (int i = 0; i < CLD; ++i) cf[i] = 0.f;
  int c0 = cat[b*3+0], c1 = cat[b*3+1], c2 = cat[b*3+2];
  if (c0 >= 0 && c0 < 2) cf[c0] = 1.f;
  if (c1 >= 0 && c1 < 3) cf[2+c1] = 1.f;
  if (c2 >= 0 && c2 < 4) cf[5+c2] = 1.f;
  for (int i = 0; i < CLD; ++i) {
    catfeat[(size_t)b*CLD + i] = cf[i];
    hcat[(size_t)b*HSZ + 192 + i] = cf[i];
  }
  for (int c = 0; c < CED; ++c) {
    float acc = bc[c];
    for (int i = 0; i < CLD; ++i) acc += cf[i]*Wc[c*CLD + i];
    catembed[(size_t)b*CED + c] = acc;
  }
}

// ---------------- pairwise 4-vector kinematics --------------------------------
__global__ void pairwise_kernel(const float* __restrict__ x4, int Nx,
                                const float* __restrict__ y4, int Ny,
                                float* __restrict__ feats, float* __restrict__ edgeP4,
                                int B)
{
  int idx = blockIdx.x * blockDim.x + threadIdx.x;
  int total = B * Nx * Ny;
  if (idx >= total) return;
  int j = idx % Ny;
  int i = (idx / Ny) % Nx;
  int b = idx / (Nx * Ny);
  const float* xp = x4 + ((size_t)b*Nx + i)*4;
  const float* yp = y4 + ((size_t)b*Ny + j)*4;
  float pTx = xp[0], ex = xp[1], phx = xp[2], mx = xp[3];
  float pTy = yp[0], ey = yp[1], phy = yp[2], my = yp[3];
  float pxx = pTx*cosf(phx), pyx = pTx*sinf(phx), pzx = pTx*sinhf(ex);
  float Ex  = sqrtf(pTx*pTx + pzx*pzx + mx*mx);
  float pxy = pTy*cosf(phy), pyy = pTy*sinf(phy), pzy = pTy*sinhf(ey);
  float Ey  = sqrtf(pTy*pTy + pzy*pzy + my*my);
  float Es = Ex + Ey, pxs = pxx + pxy, pys = pyx + pyy, pzs = pzx + pzy;
  float p2 = pxs*pxs + pys*pys + pzs*pzs;
  float m2 = fmaxf(Es*Es - p2, 0.f);
  float ppT = sqrtf(pxs*pxs + pys*pys + EPSF);
  float ceta = atanhf(pzs / sqrtf(p2 + EPSF));
  if (__builtin_isinf(ceta)) ceta = 20.f;
  float cphi = atan2f(pys, pxs);
  bool nmx = (pTx==0.f)&&(ex==0.f)&&(phx==0.f)&&(mx==0.f);
  bool nmy = (pTy==0.f)&&(ey==0.f)&&(phy==0.f)&&(my==0.f);
  bool em = nmx || nmy;
  float deta = ex - ey;
  float d = fmodf(phx - phy + PIF, 2.f*PIF);
  if (d < 0.f) d += 2.f*PIF;
  d -= PIF;
  float dist = sqrtf(deta*deta + d*d);
  float minpT = fminf(pTx, pTy);
  float kT = minpT * dist;
  float z  = minpT / (pTx + pTy);
  dist = fmaxf(dist, EPSF); kT = fmaxf(kT, EPSF); z = fmaxf(z, EPSF);
  float m2c = fmaxf(m2, EPSF);
  float ppc = fmaxf(ppT, EPSF);
  float* fo = feats + (size_t)idx*PFD;
  if (em) {
    for (int f = 0; f < PFD; ++f) fo[f] = -1e6f;
  } else {
    fo[0] = logf(dist); fo[1] = logf(kT); fo[2] = logf(z);
    fo[3] = logf(m2c);  fo[4] = logf(ppc); fo[5] = ceta; fo[6] = cphi;
  }
  if (edgeP4) {
    float* eo = edgeP4 + (size_t)idx*4;
    eo[0] = ppc; eo[1] = ceta; eo[2] = cphi; eo[3] = sqrtf(m2c);
  }
}

// ---------------- upper-triangular gathers -----------------------------------
__global__ void ut_gather_jet_kernel(const float* __restrict__ jedge,
                                     const float* __restrict__ jedgeP4,
                                     float* __restrict__ jjf, float* __restrict__ jjP4, int B)
{
  int idx = blockIdx.x * blockDim.x + threadIdx.x;
  if (idx >= B * NUTJ) return;
  int k = idx % NUTJ;
  int b = idx / NUTJ;
  int i = 0, kk = k;
  while (kk >= NJET - 1 - i) { kk -= NJET - 1 - i; ++i; }
  int j = i + 1 + kk;
  size_t src = ((size_t)b*NJET + i)*NJET + j;
  for (int f = 0; f < PFD; ++f) jjf[(size_t)idx*PFD + f] = jedge[src*PFD + f];
  for (int c = 0; c < 4;   ++c) jjP4[(size_t)idx*4   + c] = jedgeP4[src*4 + c];
}

__global__ void ut_gather_lep_kernel(const float* __restrict__ ledge,
                                     const float* __restrict__ ledgeP4,
                                     float* __restrict__ elf, float* __restrict__ llP4, int B)
{
  int b = blockIdx.x * blockDim.x + threadIdx.x;
  if (b >= B) return;
  size_t src = (size_t)b*4 + 1;   // (i=0, j=1) of 2x2 grid
  for (int f = 0; f < PFD; ++f) elf[(size_t)b*PFD + f] = ledge[src*PFD + f];
  for (int c = 0; c < 4;   ++c) llP4[(size_t)b*4   + c] = ledgeP4[src*4 + c];
}

// ---------------- conv1d attention bias over pairwise feats -------------------
__global__ void convbias_kernel(const float* __restrict__ jedge,
                                const float* __restrict__ cw, const float* __restrict__ cb,
                                float* __restrict__ biasout, int B)
{
  int idx = blockIdx.x * blockDim.x + threadIdx.x;
  int total = B * NHEAD * NJET * NJET;
  if (idx >= total) return;
  int j = idx % NJET;
  int i = (idx / NJET) % NJET;
  int h = (idx / (NJET*NJET)) % NHEAD;
  int b = idx / (NHEAD*NJET*NJET);
  const float* fe = jedge + (((size_t)b*NJET + i)*NJET + j)*PFD;
  float acc = cb[h];
  for (int f = 0; f < PFD; ++f) acc += fe[f]*cw[h*PFD + f];
  biasout[idx] = acc;
}

// ---------------- generic token embedding: concat(A, Bf, catembed) @ W^T + b --
__global__ void embed_kernel(const float* __restrict__ A, int FA,
                             const float* __restrict__ Bf, int FB,
                             const float* __restrict__ ce,
                             const float* __restrict__ W, const float* __restrict__ bias,
                             float* __restrict__ out, int L, int E,
                             int outStride, int outOff, int B)
{
  int idx = blockIdx.x * blockDim.x + threadIdx.x;
  int total = B * L * E;
  if (idx >= total) return;
  int e = idx % E;
  int t = (idx / E) % L;
  int b = idx / (L * E);
  int F = FA + FB + CED;
  const float* w = W + (size_t)e*F;
  float acc = bias[e];
  const float* ap = A + ((size_t)b*L + t)*FA;
  for (int k = 0; k < FA; ++k) acc += ap[k]*w[k];
  if (FB > 0) {
    const float* bp = Bf + ((size_t)b*L + t)*FB;
    for (int k = 0; k < FB; ++k) acc += bp[k]*w[FA + k];
  }
  acc += ce[(size_t)b*CED + 0]*w[FA+FB] + ce[(size_t)b*CED + 1]*w[FA+FB+1];
  out[(size_t)b*outStride + outOff + t*E + e] = acc;
}

// ---------------- fused MHA + LayerNorm + mean-pool (one block per sample) ----
__global__ void mha_ln_pool_kernel(const float* __restrict__ x, int L, int E,
                                   const float* __restrict__ Wqkv, const float* __restrict__ bqkv,
                                   const float* __restrict__ Wo, const float* __restrict__ bo,
                                   const float* __restrict__ lng, const float* __restrict__ lnb,
                                   const float* __restrict__ attbias,
                                   float* __restrict__ hcat, int outOff)
{
  extern __shared__ float sm[];
  int b = blockIdx.x;
  int tid = threadIdx.x, NT = blockDim.x;
  int E3 = 3*E;
  float* sx = sm;             // L*E
  float* sq = sx + L*E;       // L*3E (qkv), later reused for proj/LN
  float* ss = sq + (size_t)L*E3;   // L*L
  float* so = ss + (size_t)L*L;    // L*E
  for (int i = tid; i < L*E; i += NT) sx[i] = x[(size_t)b*L*E + i];
  __syncthreads();
  for (int i = tid; i < L*E3; i += NT) {
    int t = i / E3, o = i - t*E3;
    const float* w = Wqkv + (size_t)o*E;
    const float* xv = sx + t*E;
    float acc = bqkv[o];
    for (int k = 0; k < E; ++k) acc += xv[k]*w[k];
    sq[i] = acc;
  }
  for (int i = tid; i < L*E; i += NT) so[i] = 0.f;
  __syncthreads();
  int dh = E >> 2;
  float scale = rsqrtf((float)dh);
  for (int h = 0; h < NHEAD; ++h) {
    for (int i = tid; i < L*L; i += NT) {
      int r = i / L, c = i - r*L;
      const float* qp = sq + r*E3 + h*dh;
      const float* kp = sq + c*E3 + E + h*dh;
      float acc = 0.f;
      for (int d = 0; d < dh; ++d) acc += qp[d]*kp[d];
      acc *= scale;
      if (attbias) acc += attbias[(((size_t)b*NHEAD + h)*L + r)*L + c];
      ss[i] = acc;
    }
    __syncthreads();
    for (int r = tid; r < L; r += NT) {
      float* row = ss + r*L;
      float mxv = row[0];
      for (int c = 1; c < L; ++c) mxv = fmaxf(mxv, row[c]);
      float sum = 0.f;
      for (int c = 0; c < L; ++c) { float e = expf(row[c]-mxv); row[c] = e; sum += e; }
      float inv = 1.f/sum;
      for (int c = 0; c < L; ++c) row[c] *= inv;
    }
    __syncthreads();
    for (int i = tid; i < L*dh; i += NT) {
      int r = i / dh, d = i - r*dh;
      const float* row = ss + r*L;
      float acc = 0.f;
      for (int c = 0; c < L; ++c) acc += row[c]*sq[c*E3 + 2*E + h*dh + d];
      so[r*E + h*dh + d] = acc;
    }
    __syncthreads();
  }
  // output projection (reuse sq[0..L*E))
  for (int i = tid; i < L*E; i += NT) {
    int t = i / E, e = i - t*E;
    const float* w = Wo + (size_t)e*E;
    const float* ov = so + t*E;
    float acc = bo[e];
    for (int k = 0; k < E; ++k) acc += ov[k]*w[k];
    sq[i] = acc;
  }
  __syncthreads();
  // LayerNorm per token
  for (int t = tid; t < L; t += NT) {
    float* row = sq + t*E;
    float m = 0.f; for (int e = 0; e < E; ++e) m += row[e];
    m /= (float)E;
    float v = 0.f; for (int e = 0; e < E; ++e) { float dd = row[e]-m; v += dd*dd; }
    v /= (float)E;
    float inv = rsqrtf(v + 1e-5f);
    for (int e = 0; e < E; ++e) row[e] = (row[e]-m)*inv*lng[e] + lnb[e];
  }
  __syncthreads();
  // mean over tokens
  for (int e = tid; e < E; e += NT) {
    float acc = 0.f;
    for (int t = 0; t < L; ++t) acc += sq[t*E + e];
    hcat[(size_t)b*HSZ + outOff + e] = acc / (float)L;
  }
}

// ---------------- f32 -> padded f16 row packer --------------------------------
__global__ void pack_f16_kernel(const float* __restrict__ a, int Ka,
                                const float* __restrict__ bsrc, int Kb,
                                _Float16* __restrict__ out, int Kpad, int B)
{
  int idx = blockIdx.x * blockDim.x + threadIdx.x;
  int total = B * Kpad;
  if (idx >= total) return;
  int c = idx % Kpad;
  int r = idx / Kpad;
  float v = 0.f;
  if (c < Ka) v = a[(size_t)r*Ka + c];
  else if (c < Ka + Kb) v = bsrc[(size_t)r*Kb + (c - Ka)];
  out[idx] = (_Float16)v;
}

__global__ void convert_w_kernel(const float* __restrict__ W, _Float16* __restrict__ out,
                                 int N, int K, int Kpad, int Npad)
{
  int idx = blockIdx.x * blockDim.x + threadIdx.x;
  int total = Npad * Kpad;
  if (idx >= total) return;
  int c = idx % Kpad;
  int n = idx / Kpad;
  float v = (n < N && c < K) ? W[(size_t)n*K + c] : 0.f;
  out[idx] = (_Float16)v;
}

// ---------------- FC layer via v_wmma_f32_16x16x32_f16 ------------------------
// One wave per 16x16 output tile. A = activations [B,Kpad] f16, B = W[n][k] f16.
__global__ void fc_wmma_kernel(const _Float16* __restrict__ X,
                               const _Float16* __restrict__ Wh,
                               const float* __restrict__ fb,
                               const float* __restrict__ bng, const float* __restrict__ bnb,
                               const float* __restrict__ bnm, const float* __restrict__ bnv,
                               float* __restrict__ out, int Kpad, int Nout, int outLd, int relu)
{
  int mt = blockIdx.x, nt = blockIdx.y;
  int lane = threadIdx.x;
  int lo = lane & 15;
  int hi = lane >> 4;
  const _Float16* arow = X  + (size_t)(mt*16 + lo)*Kpad;
  const _Float16* brow = Wh + (size_t)(nt*16 + lo)*Kpad;
  v8f acc = {};
  union HV { v16h v; v8h h[2]; };
  HV A, Bm;
  int ksteps = Kpad >> 5;
  for (int kt = 0; kt < ksteps; ++kt) {
    int kb = kt << 5;
    // A 16x32 f16: lane<16 -> K {0..7,16..23}; lane>=16 -> K {8..15,24..31}
    int a0 = kb + (hi ? 8 : 0);
    A.h[0] = *(const v8h*)(arow + a0);
    A.h[1] = *(const v8h*)(arow + a0 + 16);
    // B 32x16 f16: lane<16 -> K {0..7,16..23}; lane>=16 -> K {8..15,24..31}; N = lane%16
    int b0 = kb + (hi ? 8 : 0);
    Bm.h[0] = *(const v8h*)(brow + b0);
    Bm.h[1] = *(const v8h*)(brow + b0 + 16);
    acc = __builtin_amdgcn_wmma_f32_16x16x32_f16(false, A.v, false, Bm.v,
                                                 (short)0, acc, false, false);
  }
  int col = nt*16 + lo;
  if (col >= Nout) return;
  float bb = fb[col];
  float g = 1.f, be = 0.f, mu = 0.f, istd = 1.f;
  if (bng) { g = bng[col]; be = bnb[col]; mu = bnm[col]; istd = rsqrtf(bnv[col] + 1e-5f); }
  for (int v = 0; v < 8; ++v) {
    int row = mt*16 + v + (hi ? 8 : 0);   // C layout: VGPR v -> M=v (+8 for hi lanes)
    float val = acc[v] + bb;
    if (bng) val = (val - mu)*istd*g + be;
    if (relu) val = fmaxf(val, 0.f);
    out[(size_t)row*outLd + col] = val;
  }
}

// =============================================================================
extern "C" void kernel_launch(void* const* d_in, const int* in_sizes, int n_in,
                              void* d_out, int out_size, void* d_ws, size_t ws_size,
                              hipStream_t stream)
{
  (void)in_sizes; (void)n_in; (void)out_size; (void)ws_size;
  const int B = BATCH;
  const float* jet   = (const float*)d_in[0];
  const float* jetp4 = (const float*)d_in[1];
  const float* lep   = (const float*)d_in[2];
  const float* lepp4 = (const float*)d_in[3];
  const float* ll    = (const float*)d_in[4];
  const float* glo   = (const float*)d_in[5];
  const int*   cat   = (const int*)d_in[6];
#define PR(i) ((const float*)d_in[(i)])

  // ---- workspace carve-out (256B aligned chunks) ----
  size_t woff = 0;
  auto alloc = [&](size_t bytes) -> char* {
    char* r = (char*)d_ws + woff;
    woff += (bytes + 255) & ~(size_t)255;
    return r;
  };
  float* catfeat  = (float*)alloc((size_t)B*CLD*4);
  float* catembed = (float*)alloc((size_t)B*CED*4);
  float* jedge    = (float*)alloc((size_t)B*NJET*NJET*PFD*4);
  float* jedgeP4  = (float*)alloc((size_t)B*NJET*NJET*4*4);
  float* ledge    = (float*)alloc((size_t)B*NLEP*NLEP*PFD*4);
  float* ledgeP4  = (float*)alloc((size_t)B*NLEP*NLEP*4*4);
  float* jjf      = (float*)alloc((size_t)B*NUTJ*PFD*4);
  float* jjP4     = (float*)alloc((size_t)B*NUTJ*4*4);
  float* elf      = (float*)alloc((size_t)B*PFD*4);
  float* llP4     = (float*)alloc((size_t)B*4*4);
  float* jlf      = (float*)alloc((size_t)B*NJET*NLEP*PFD*4);
  float* jjlf     = (float*)alloc((size_t)B*NUTJ*NLEP*PFD*4);
  float* jllf     = (float*)alloc((size_t)B*NJET*PFD*4);
  float* jjllf    = (float*)alloc((size_t)B*NUTJ*PFD*4);
  float* abias    = (float*)alloc((size_t)B*NHEAD*NJET*NJET*4);
  float* jemb     = (float*)alloc((size_t)B*NJET*E2D*4);
  float* ejee     = (float*)alloc((size_t)B*NUTJ*E2D*4);
  float* lemb     = (float*)alloc((size_t)B*NLEP*E1D*4);
  float* elee     = (float*)alloc((size_t)B*E1D*4);
  float* jle      = (float*)alloc((size_t)B*20*E1D*4);
  float* jjle     = (float*)alloc((size_t)B*90*E1D*4);
  float* jlle     = (float*)alloc((size_t)B*10*E1D*4);
  float* jjlle    = (float*)alloc((size_t)B*45*E1D*4);
  float* hcat     = (float*)alloc((size_t)B*HSZ*4);
  _Float16* X1    = (_Float16*)alloc((size_t)B*K1PAD*2);
  _Float16* X2    = (_Float16*)alloc((size_t)B*K2PAD*2);
  _Float16* X3    = (_Float16*)alloc((size_t)B*K2PAD*2);
  _Float16* X4    = (_Float16*)alloc((size_t)B*K2PAD*2);
  float* a1       = (float*)alloc((size_t)B*ADIM*4);
  float* a2       = (float*)alloc((size_t)B*ADIM*4);
  float* a3       = (float*)alloc((size_t)B*ADIM*4);
  _Float16* W1h   = (_Float16*)alloc((size_t)ADIM*K1PAD*2);
  _Float16* W2h   = (_Float16*)alloc((size_t)ADIM*K2PAD*2);
  _Float16* W3h   = (_Float16*)alloc((size_t)ADIM*K2PAD*2);
  _Float16* W4h   = (_Float16*)alloc((size_t)16*K2PAD*2);

  dim3 blk(256);
  auto grid = [](long total) { return dim3((unsigned)((total + 255)/256)); };

  // ---- categorical features + embedding ----
  catfeat_kernel<<<grid(B), blk, 0, stream>>>(cat, PR(P_EMBEDCAT_W), PR(P_EMBEDCAT_B),
                                              catfeat, catembed, hcat, B);
  // ---- pairwise kinematics ----
  pairwise_kernel<<<grid((long)B*NJET*NJET), blk, 0, stream>>>(jetp4, NJET, jetp4, NJET, jedge, jedgeP4, B);
  pairwise_kernel<<<grid((long)B*NLEP*NLEP), blk, 0, stream>>>(lepp4, NLEP, lepp4, NLEP, ledge, ledgeP4, B);
  ut_gather_jet_kernel<<<grid((long)B*NUTJ), blk, 0, stream>>>(jedge, jedgeP4, jjf, jjP4, B);
  ut_gather_lep_kernel<<<grid(B), blk, 0, stream>>>(ledge, ledgeP4, elf, llP4, B);
  pairwise_kernel<<<grid((long)B*NJET*NLEP), blk, 0, stream>>>(jetp4, NJET, lepp4, NLEP, jlf, nullptr, B);
  pairwise_kernel<<<grid((long)B*NUTJ*NLEP), blk, 0, stream>>>(jjP4, NUTJ, lepp4, NLEP, jjlf, nullptr, B);
  pairwise_kernel<<<grid((long)B*NJET), blk, 0, stream>>>(jetp4, NJET, llP4, 1, jllf, nullptr, B);
  pairwise_kernel<<<grid((long)B*NUTJ), blk, 0, stream>>>(jjP4, NUTJ, llP4, 1, jjllf, nullptr, B);
  convbias_kernel<<<grid((long)B*NHEAD*NJET*NJET), blk, 0, stream>>>(jedge, PR(P_CONV_W), PR(P_CONV_B), abias, B);

  // ---- token embeddings ----
  embed_kernel<<<grid((long)B*NJET*E2D), blk, 0, stream>>>(jet, 2, jetp4, 4, catembed,
      PR(P_EMBEDJETS_W), PR(P_EMBEDJETS_B), jemb, NJET, E2D, NJET*E2D, 0, B);
  embed_kernel<<<grid((long)B*NUTJ*E2D), blk, 0, stream>>>(jjf, PFD, nullptr, 0, catembed,
      PR(P_EDGEJET_W), PR(P_EDGEJET_B), ejee, NUTJ, E2D, NUTJ*E2D, 0, B);
  embed_kernel<<<grid((long)B*NLEP*E1D), blk, 0, stream>>>(lep, 2, lepp4, 4, catembed,
      PR(P_EMBEDLEPS_W), PR(P_EMBEDLEPS_B), lemb, NLEP, E1D, NLEP*E1D, 0, B);
  embed_kernel<<<grid((long)B*E1D), blk, 0, stream>>>(elf, PFD, nullptr, 0, catembed,
      PR(P_EDGELEP_W), PR(P_EDGELEP_B), elee, 1, E1D, E1D, 0, B);
  embed_kernel<<<grid((long)B*20*E1D), blk, 0, stream>>>(jlf, PFD, nullptr, 0, catembed,
      PR(P_EMBEDJL_W), PR(P_EMBEDJL_B), jle, 20, E1D, 20*E1D, 0, B);
  embed_kernel<<<grid((long)B*90*E1D), blk, 0, stream>>>(jjlf, PFD, nullptr, 0, catembed,
      PR(P_EMBEDJJL_W), PR(P_EMBEDJJL_B), jjle, 90, E1D, 90*E1D, 0, B);
  embed_kernel<<<grid((long)B*10*E1D), blk, 0, stream>>>(jllf, PFD, nullptr, 0, catembed,
      PR(P_EMBEDJLL_W), PR(P_EMBEDJLL_B), jlle, 10, E1D, 10*E1D, 0, B);
  embed_kernel<<<grid((long)B*45*E1D), blk, 0, stream>>>(jjllf, PFD, nullptr, 0, catembed,
      PR(P_EMBEDJJLL_W), PR(P_EMBEDJJLL_B), jjlle, 45, E1D, 45*E1D, 0, B);
  // llv and glob are plain linears written straight into hcat
  embed_kernel<<<grid((long)B*E1D), blk, 0, stream>>>(ll, 4, nullptr, 0, catembed,
      PR(P_EMBEDLL_W), PR(P_EMBEDLL_B), hcat, 1, E1D, HSZ, 96, B);
  embed_kernel<<<grid((long)B*E1D), blk, 0, stream>>>(glo, 4, nullptr, 0, catembed,
      PR(P_EMBEDGLOB_W), PR(P_EMBEDGLOB_B), hcat, 1, E1D, HSZ, 176, B);

  // ---- fused MHA + LN + mean-pool branches (one block per sample) ----
  auto smemB = [](int L, int E) { return (size_t)(5*L*E + L*L)*4; };
  mha_ln_pool_kernel<<<B, 128, smemB(NJET, E2D), stream>>>(jemb, NJET, E2D,
      PR(P_MHAJET_WQKV), PR(P_MHAJET_BQKV), PR(P_MHAJET_WO), PR(P_MHAJET_BO),
      PR(P_LN2X_G), PR(P_LN2X_B), abias, hcat, 0);
  mha_ln_pool_kernel<<<B, 128, smemB(NUTJ, E2D), stream>>>(ejee, NUTJ, E2D,
      PR(P_MHAEJ_WQKV), PR(P_MHAEJ_BQKV), PR(P_MHAEJ_WO), PR(P_MHAEJ_BO),
      PR(P_LN2X_G), PR(P_LN2X_B), nullptr, hcat, 32);
  mha_ln_pool_kernel<<<B, 128, smemB(NLEP, E1D), stream>>>(lemb, NLEP, E1D,
      PR(P_MHALEP_WQKV), PR(P_MHALEP_BQKV), PR(P_MHALEP_WO), PR(P_MHALEP_BO),
      PR(P_LN_G), PR(P_LN_B), nullptr, hcat, 64);
  mha_ln_pool_kernel<<<B, 128, smemB(1, E1D), stream>>>(elee, 1, E1D,
      PR(P_MHAEL_WQKV), PR(P_MHAEL_BQKV), PR(P_MHAEL_WO), PR(P_MHAEL_BO),
      PR(P_LN_G), PR(P_LN_B), nullptr, hcat, 80);
  mha_ln_pool_kernel<<<B, 128, smemB(20, E1D), stream>>>(jle, 20, E1D,
      PR(P_MHAJL_WQKV), PR(P_MHAJL_BQKV), PR(P_MHAJL_WO), PR(P_MHAJL_BO),
      PR(P_LN_G), PR(P_LN_B), nullptr, hcat, 112);
  mha_ln_pool_kernel<<<B, 128, smemB(90, E1D), stream>>>(jjle, 90, E1D,
      PR(P_MHAJJL_WQKV), PR(P_MHAJJL_BQKV), PR(P_MHAJJL_WO), PR(P_MHAJJL_BO),
      PR(P_LN_G), PR(P_LN_B), nullptr, hcat, 128);
  mha_ln_pool_kernel<<<B, 128, smemB(10, E1D), stream>>>(jlle, 10, E1D,
      PR(P_MHAJLL_WQKV), PR(P_MHAJLL_BQKV), PR(P_MHAJLL_WO), PR(P_MHAJLL_BO),
      PR(P_LN_G), PR(P_LN_B), nullptr, hcat, 144);
  mha_ln_pool_kernel<<<B, 128, smemB(45, E1D), stream>>>(jjlle, 45, E1D,
      PR(P_MHAJJLL_WQKV), PR(P_MHAJJLL_BQKV), PR(P_MHAJJLL_WO), PR(P_MHAJJLL_BO),
      PR(P_LN_G), PR(P_LN_B), nullptr, hcat, 160);

  // ---- classifier head: WMMA f16 GEMMs with BN/ReLU epilogues ----
  convert_w_kernel<<<grid((long)ADIM*K1PAD), blk, 0, stream>>>(PR(P_FC1_W), W1h, ADIM, HSZ, K1PAD, ADIM);
  convert_w_kernel<<<grid((long)ADIM*K2PAD), blk, 0, stream>>>(PR(P_FC2_W), W2h, ADIM, ADIM+CLD, K2PAD, ADIM);
  convert_w_kernel<<<grid((long)ADIM*K2PAD), blk, 0, stream>>>(PR(P_FC3_W), W3h, ADIM, ADIM+CLD, K2PAD, ADIM);
  convert_w_kernel<<<grid((long)16*K2PAD), blk, 0, stream>>>(PR(P_FC4_W), W4h, 2, ADIM+CLD, K2PAD, 16);

  pack_f16_kernel<<<grid((long)B*K1PAD), blk, 0, stream>>>(hcat, HSZ, nullptr, 0, X1, K1PAD, B);
  fc_wmma_kernel<<<dim3(B/16, ADIM/16), dim3(32), 0, stream>>>(X1, W1h, PR(P_FC1_B),
      PR(P_BN1_G), PR(P_BN1_B), PR(P_BN1_M), PR(P_BN1_V), a1, K1PAD, ADIM, ADIM, 1);
  pack_f16_kernel<<<grid((long)B*K2PAD), blk, 0, stream>>>(a1, ADIM, catfeat, CLD, X2, K2PAD, B);
  fc_wmma_kernel<<<dim3(B/16, ADIM/16), dim3(32), 0, stream>>>(X2, W2h, PR(P_FC2_B),
      PR(P_BN2_G), PR(P_BN2_B), PR(P_BN2_M), PR(P_BN2_V), a2, K2PAD, ADIM, ADIM, 1);
  pack_f16_kernel<<<grid((long)B*K2PAD), blk, 0, stream>>>(a2, ADIM, catfeat, CLD, X3, K2PAD, B);
  fc_wmma_kernel<<<dim3(B/16, ADIM/16), dim3(32), 0, stream>>>(X3, W3h, PR(P_FC3_B),
      PR(P_BN3_G), PR(P_BN3_B), PR(P_BN3_M), PR(P_BN3_V), a3, K2PAD, ADIM, ADIM, 1);
  pack_f16_kernel<<<grid((long)B*K2PAD), blk, 0, stream>>>(a3, ADIM, catfeat, CLD, X4, K2PAD, B);
  fc_wmma_kernel<<<dim3(B/16, 1), dim3(32), 0, stream>>>(X4, W4h, PR(P_FC4_B),
      nullptr, nullptr, nullptr, nullptr, (float*)d_out, K2PAD, 2, 2, 0);
#undef PR
}
